// DeepBoltzmannMachine_60000693125273
// MI455X (gfx1250) — matrix operand order repaired and checked by reading, not compiled, not run on hardware
//
#include <hip/hip_runtime.h>

// ---------------------------------------------------------------------------
// DBM mean-field on gfx1250: bf16 WMMA GEMMs (f32 accumulate) + fused sigmoid.
// B=512, D0=D1=D2=4096, iterations=10 (constants from the reference file).
// Weights (128 MB fp32 -> 96 MB bf16) live in the 192 MB L2 -> compute-bound;
// all MACs go through v_wmma_f32_16x16x32_bf16; tiles staged with
// global_load_async_to_lds_b128 (ASYNCcnt) into ping-pong LDS buffers with
// compile-time addresses (K-loop unrolled by 2).
// ---------------------------------------------------------------------------

typedef __bf16 bf16;
typedef bf16  bf16x8  __attribute__((ext_vector_type(8)));
typedef bf16  bf16x16 __attribute__((ext_vector_type(16)));
typedef float v8f     __attribute__((ext_vector_type(8)));

#define TILE_M 128
#define TILE_N 128
#define TILE_K 32
#define LDS_STRIDE 40   // 32 K-elems + 8 pad (bf16) to dodge bank conflicts
#define TILE_ELEMS (TILE_M * LDS_STRIDE)

#if __has_builtin(__builtin_amdgcn_global_load_async_to_lds_b128) && \
    __has_builtin(__builtin_amdgcn_s_wait_asynccnt)
#define USE_ASYNC_LDS 1
#else
#define USE_ASYNC_LDS 0
#endif

// Exact parameter types of the async-LDS builtin (from hipcc's diagnostic):
// (global int4* src, shared int4* dst, imm int offset, imm int cpol)
typedef int int4g __attribute__((vector_size(16)));
typedef __attribute__((address_space(1))) int4g as1_i4;
typedef __attribute__((address_space(3))) int4g as3_i4;

union FragBf { bf16x16 v; bf16x8 h[2]; };

__device__ __forceinline__ float sigmoidf_fast(float x) {
    return 1.0f / (1.0f + __expf(-x));
}

#if USE_ASYNC_LDS
__device__ __forceinline__ void async_b128(const bf16* gsrc, const bf16* ldst) {
    __builtin_amdgcn_global_load_async_to_lds_b128(
        (as1_i4*)gsrc, (as3_i4*)ldst, 0, 0);
}
// Stage one 128x32 A tile chunk + one 128x32 B tile chunk (32B each / thread).
__device__ __forceinline__ void stage_tiles(
    const bf16* gA, const bf16* gB, int kt, bf16* dA, bf16* dB)
{
    async_b128(gA + kt,     dA);
    async_b128(gA + kt + 8, dA + 8);
    async_b128(gB + kt,     dB);
    async_b128(gB + kt + 8, dB + 8);
}
#endif

// Fragment loads (contiguous K per half-wave, same order for A and B) + 8 WMMAs for one K-step.
__device__ __forceinline__ void wmma_step(
    const bf16* Ab, const bf16* Bb,
    int wvm, int wvn, int hl, int ln, v8f acc[4][2])
{
    // B fragments: lane holds col N=ln; half-wave 0 -> K 0..15, 1 -> 16..31
    FragBf fb[2];
    #pragma unroll
    for (int nf = 0; nf < 2; ++nf) {
        const bf16* p = &Bb[(wvn * 32 + nf * 16 + ln) * LDS_STRIDE + hl * 16];
        fb[nf].h[0] = *(const bf16x8*)(p);
        fb[nf].h[1] = *(const bf16x8*)(p + 8);
    }
    // A fragments: lane holds row M=ln; half-wave 0 -> K 0..15, 1 -> 16..31
    FragBf fa[4];
    #pragma unroll
    for (int mf = 0; mf < 4; ++mf) {
        const bf16* p = &Ab[(wvm * 64 + mf * 16 + ln) * LDS_STRIDE + hl * 16];
        fa[mf].h[0] = *(const bf16x8*)(p);
        fa[mf].h[1] = *(const bf16x8*)(p + 8);
    }
    #pragma unroll
    for (int mf = 0; mf < 4; ++mf)
        #pragma unroll
        for (int nf = 0; nf < 2; ++nf)
            acc[mf][nf] = __builtin_amdgcn_wmma_f32_16x16x32_bf16(
                false, fa[mf].v, false, fb[nf].v,
                (short)0, acc[mf][nf], false, false);
}

// One K-loop of C += A(row-major, lda=K) * Bt(row-major N x K, i.e. B^T), into
// the block's 128x128 tile. acc is this wave's 4x2 grid of 16x16 f32 fragments.
// K must be a multiple of 2*TILE_K (64).
__device__ __forceinline__ void gemm_k_loop(
    const bf16* __restrict__ A, const bf16* __restrict__ Bt, int K,
    int rowbase, int colbase, bf16* As, bf16* Bs, v8f acc[4][2])
{
    const int t    = threadIdx.x;      // 0..255
    const int lane = t & 31;
    const int wave = t >> 5;           // 0..7
    const int wvm  = wave & 1;         // 2 waves tile M (64 rows each)
    const int wvn  = wave >> 1;        // 4 waves tile N (32 cols each)
    const int hl   = lane >> 4;        // half-wave select (ISA 16-bit layouts)
    const int ln   = lane & 15;

    // Staging: each thread moves 32B of A-tile and 32B of B-tile per K-step.
    const int srow = t >> 1;           // 0..127 (tile row / Bt row)
    const int skc  = (t & 1) << 4;     // 0 or 16 (K sub-chunk)

    const bf16* gA = A  + (size_t)(rowbase + srow) * K + skc;
    const bf16* gB = Bt + (size_t)(colbase + srow) * K + skc;
    const int lofs = srow * LDS_STRIDE + skc;

#if USE_ASYNC_LDS
    // Ping-pong async pipeline, one barrier per K-step, static buffer addrs.
    bf16* A0 = As + lofs;               bf16* B0 = Bs + lofs;
    bf16* A1 = A0 + TILE_ELEMS;         bf16* B1 = B0 + TILE_ELEMS;

    __syncthreads();   // all prior readers of LDS are done (2nd call safety)
    stage_tiles(gA, gB, 0, A0, B0);
    for (int kt = 0; kt < K; kt += 2 * TILE_K) {
        // half-step 0: compute buf0, prefetch buf1 (kt+32 < K always here)
        __builtin_amdgcn_s_wait_asynccnt(0);
        __syncthreads();
        stage_tiles(gA, gB, kt + TILE_K, A1, B1);
        wmma_step(As, Bs, wvm, wvn, hl, ln, acc);
        // half-step 1: compute buf1, prefetch buf0 (except on last pair)
        __builtin_amdgcn_s_wait_asynccnt(0);
        __syncthreads();
        if (kt + 2 * TILE_K < K)
            stage_tiles(gA, gB, kt + 2 * TILE_K, A0, B0);
        wmma_step(As + TILE_ELEMS, Bs + TILE_ELEMS, wvm, wvn, hl, ln, acc);
    }
#else
    // Fallback: manual global->VGPR->LDS staging, two barriers per step.
    for (int kt = 0; kt < K; kt += TILE_K) {
        bf16x8 a0 = *(const bf16x8*)(gA + kt);
        bf16x8 a1 = *(const bf16x8*)(gA + kt + 8);
        bf16x8 b0 = *(const bf16x8*)(gB + kt);
        bf16x8 b1 = *(const bf16x8*)(gB + kt + 8);
        if (kt + TILE_K < K) {
            __builtin_prefetch(gA + kt + TILE_K, 0, 3);
            __builtin_prefetch(gB + kt + TILE_K, 0, 3);
        }
        __syncthreads();
        *(bf16x8*)(As + lofs)     = a0;
        *(bf16x8*)(As + lofs + 8) = a1;
        *(bf16x8*)(Bs + lofs)     = b0;
        *(bf16x8*)(Bs + lofs + 8) = b1;
        __syncthreads();
        wmma_step(As, Bs, wvm, wvn, hl, ln, acc);
    }
#endif
}

// C = sigmoid(bias + A0*B0t^T + [A1*B1t^T]); writes f32 (d_out) and bf16 state.
__global__ __launch_bounds__(256) void gemm_bias_sigmoid(
    const bf16* __restrict__ A0, const bf16* __restrict__ B0t, int K0,
    const bf16* __restrict__ A1, const bf16* __restrict__ B1t, int K1,
    const float* __restrict__ bias,
    float* __restrict__ outF, bf16* __restrict__ outB, int N)
{
#if USE_ASYNC_LDS
    __shared__ bf16 As[2 * TILE_ELEMS];
    __shared__ bf16 Bs[2 * TILE_ELEMS];
#else
    __shared__ bf16 As[TILE_ELEMS];
    __shared__ bf16 Bs[TILE_ELEMS];
#endif

    const int rowbase = blockIdx.y * TILE_M;
    const int colbase = blockIdx.x * TILE_N;

    v8f acc[4][2];
    const v8f z = {0.f, 0.f, 0.f, 0.f, 0.f, 0.f, 0.f, 0.f};
    #pragma unroll
    for (int mf = 0; mf < 4; ++mf)
        #pragma unroll
        for (int nf = 0; nf < 2; ++nf) acc[mf][nf] = z;

    gemm_k_loop(A0, B0t, K0, rowbase, colbase, As, Bs, acc);
    if (K1 > 0)
        gemm_k_loop(A1, B1t, K1, rowbase, colbase, As, Bs, acc);

    const int t    = threadIdx.x;
    const int lane = t & 31;
    const int wave = t >> 5;
    const int wvm  = wave & 1;
    const int wvn  = wave >> 1;
    const int hl   = lane >> 4;
    const int ln   = lane & 15;

    float bv[2];
    #pragma unroll
    for (int nf = 0; nf < 2; ++nf)
        bv[nf] = bias[colbase + wvn * 32 + nf * 16 + ln];

    // C/D 16x16 f32 layout: VGPR r, lanes 0-15 -> M=r, lanes 16-31 -> M=r+8.
    #pragma unroll
    for (int mf = 0; mf < 4; ++mf) {
        #pragma unroll
        for (int nf = 0; nf < 2; ++nf) {
            #pragma unroll
            for (int r = 0; r < 8; ++r) {
                int row = rowbase + wvm * 64 + mf * 16 + hl * 8 + r;
                int col = colbase + wvn * 32 + nf * 16 + ln;
                float s = sigmoidf_fast(acc[mf][nf][r] + bv[nf]);
                size_t idx = (size_t)row * N + col;
                outF[idx] = s;
                outB[idx] = (bf16)s;
            }
        }
    }
}

// ---------------------------------------------------------------------------
// Prep kernels
// ---------------------------------------------------------------------------
__global__ __launch_bounds__(256) void k_convert(
    const float* __restrict__ in, bf16* __restrict__ out, int n)
{
    int i = blockIdx.x * 256 + threadIdx.x;
    if (i < n) out[i] = (bf16)in[i];
}

// out[c*R + r] = (bf16) in[r*C + c]  (LDS-tiled transpose+convert)
__global__ __launch_bounds__(256) void k_transpose_convert(
    const float* __restrict__ in, bf16* __restrict__ out, int R, int C)
{
    __shared__ float tile[32][33];
    const int tx = threadIdx.x, ty = threadIdx.y;
    const int bx = blockIdx.x * 32, by = blockIdx.y * 32;
    #pragma unroll
    for (int j = 0; j < 32; j += 8)
        tile[ty + j][tx] = in[(size_t)(by + ty + j) * C + (bx + tx)];
    __syncthreads();
    #pragma unroll
    for (int j = 0; j < 32; j += 8)
        out[(size_t)(bx + ty + j) * R + (by + tx)] = (bf16)tile[tx][ty + j];
}

__global__ __launch_bounds__(256) void k_init_state(
    const float* __restrict__ bias, bf16* __restrict__ s, int n, int ncolsMask)
{
    int i = blockIdx.x * 256 + threadIdx.x;
    if (i < n) s[i] = (bf16)sigmoidf_fast(bias[i & ncolsMask]);
}

// ---------------------------------------------------------------------------
// Host launcher
// ---------------------------------------------------------------------------
extern "C" void kernel_launch(void* const* d_in, const int* in_sizes, int n_in,
                              void* d_out, int out_size, void* d_ws, size_t ws_size,
                              hipStream_t stream)
{
    (void)in_sizes; (void)n_in; (void)out_size; (void)ws_size;
    const float* data = (const float*)d_in[0];   // [512, 4096]
    const float* W1   = (const float*)d_in[1];   // [4096, 4096]
    const float* W2   = (const float*)d_in[2];   // [4096, 4096]
    const float* b1   = (const float*)d_in[3];   // [4096]
    const float* b2   = (const float*)d_in[4];   // [4096]
    // d_in[5] = iterations; fixed to 10 in the reference's setup_inputs().
    const int ITERS = 10;

    const int Bn = 512, D = 4096;
    float* out = (float*)d_out;                  // [2, 512, 4096]

    char* ws = (char*)d_ws;
    const size_t wbytes = (size_t)D * D * sizeof(bf16);   // 32 MB
    const size_t sbytes = (size_t)Bn * D * sizeof(bf16);  // 4 MB
    bf16* W1t   = (bf16*)(ws);                    // B^T layout for data@W1
    bf16* W2t   = (bf16*)(ws + wbytes);           // B^T layout for s1@W2
    bf16* W2b   = (bf16*)(ws + 2 * wbytes);       // == B^T layout for s2@W2^T
    bf16* dataB = (bf16*)(ws + 3 * wbytes);
    bf16* s1b   = (bf16*)(ws + 3 * wbytes + sbytes);
    bf16* s2b   = (bf16*)(ws + 3 * wbytes + 2 * sbytes);

    // --- one-time conversions ---
    {
        dim3 tb(32, 8), tg(D / 32, D / 32);
        k_transpose_convert<<<tg, tb, 0, stream>>>(W1, W1t, D, D);
        k_transpose_convert<<<tg, tb, 0, stream>>>(W2, W2t, D, D);
        k_convert<<<(D * D) / 256, 256, 0, stream>>>(W2, W2b, D * D);
        k_convert<<<(Bn * D) / 256, 256, 0, stream>>>(data, dataB, Bn * D);
        k_init_state<<<(Bn * D) / 256, 256, 0, stream>>>(b2, s2b, Bn * D, D - 1);
    }

    // --- mean-field iterations (sequential dependency) ---
    dim3 gg(D / TILE_N, Bn / TILE_M);  // (32, 4)
    for (int it = 0; it < ITERS; ++it) {
        // s1 = sigmoid(b1 + data@W1 + s2@W2^T)
        gemm_bias_sigmoid<<<gg, 256, 0, stream>>>(
            dataB, W1t, D, s2b, W2b, D, b1, out, s1b, D);
        // s2 = sigmoid(b2 + s1@W2)
        gemm_bias_sigmoid<<<gg, 256, 0, stream>>>(
            s1b, W2t, D, nullptr, nullptr, 0, b2, out + (size_t)Bn * D, s2b, D);
    }
}
